// PredRNNPPForecaster_73753178406930
// MI455X (gfx1250) — matrix operand order repaired and checked by reading, not compiled, not run on hardware
//
#include <hip/hip_runtime.h>
#include <cstddef>
#include <utility>

// ---------------- types ----------------
typedef __attribute__((ext_vector_type(16))) __bf16          v16bf;
typedef __attribute__((ext_vector_type(16))) unsigned short  v16u;
typedef __attribute__((ext_vector_type(8)))  float           v8f;
typedef __attribute__((ext_vector_type(4)))  unsigned int    v4u;

#define CH_ELEMS (4 * 64 * 4096)   // one [B=4, 64, 64, 64] fp32 tensor

__device__ __forceinline__ unsigned short f2bf_bits(float f) {
  unsigned u = __builtin_bit_cast(unsigned, f);
  unsigned r = u + 0x7FFFu + ((u >> 16) & 1u);   // round-to-nearest-even
  return (unsigned short)(r >> 16);
}
__device__ __forceinline__ float sigmoidf_(float x) { return 1.0f / (1.0f + __expf(-x)); }

__device__ __forceinline__ float at_blk(const float* __restrict__ t, int C, int blk,
                                        int b, int ch, int pix) {
  return t[((((size_t)b * C) + (size_t)blk * 64 + ch) << 12) + pix];
}

// ---------------- implicit-GEMM conv via WMMA bf16 + async LDS staging ----------------
// GEMM: M = Cout (4 waves x 16), N = one image row (4 x 16 pixels), K = Cin*KS*KS (32-step).
// Input activations are bf16 [B][Cin][64][64]; weights bf16 [Cout][Kpad32] (zero-padded).
template <int KS>
__global__ __launch_bounds__(128)
void conv_wmma_kernel(const unsigned short* __restrict__ wq,
                      const float* __restrict__ bias,
                      const unsigned short* __restrict__ inb,
                      float* __restrict__ out, int outBS,
                      int Cin, int Cout, int Ktot, int Kpad32, int accumulate) {
  constexpr int KS2 = KS * KS;
  constexpr int PAD = KS / 2;
  __shared__ __align__(16) unsigned short sh[128 * 3 * 64];   // up to 48 KB staged rows

  const int tid = threadIdx.x;
  const int row = blockIdx.y;          // b*64 + y
  const int b   = row >> 6;
  const int y   = row & 63;

  // ---- stage Cin*KS bf16 rows of the input into LDS (async global->LDS) ----
  const unsigned ldsbase = (unsigned)(size_t)(&sh[0]);
  const int nChunks = Cin * KS * 8;    // 16-byte chunks (one row = 128 B)
  for (int c = tid; c < nChunks; c += 128) {
    const int part  = c & 7;
    const int rowid = c >> 3;          // ci*KS + r
    const int ci    = rowid / KS;
    const int r     = rowid - ci * KS;
    const int iy    = y + r - PAD;
    if ((unsigned)iy < 64u) {
      const unsigned goff = ((((unsigned)b * Cin + ci) << 12) + ((unsigned)iy << 6)) * 2u
                            + (unsigned)(part * 16);
      asm volatile("global_load_async_to_lds_b128 %0, %1, %2"
                   :: "v"(ldsbase + (unsigned)rowid * 128u + (unsigned)(part * 16)),
                      "v"(goff), "s"(inb)
                   : "memory");
    } else {
      v4u z = {0u, 0u, 0u, 0u};
      *(v4u*)((char*)&sh[0] + (size_t)rowid * 128 + part * 16) = z;
    }
  }
  asm volatile("s_wait_asynccnt 0" ::: "memory");
  __syncthreads();

  // ---- per-wave 16(M) x 64(N) output tile ----
  const int wave  = tid >> 5;
  const int lane  = tid & 31;
  const int mTile = blockIdx.x * 4 + wave;
  if (mTile * 16 >= Cout) return;

  const int half   = lane >> 4;
  const int co     = mTile * 16 + (lane & 15);
  const int nloc   = lane & 15;
  const int kBhalf = half << 4;
  const bool coOK  = co < Cout;
  const unsigned short* wrow = wq + (size_t)co * Kpad32;

  v8f acc[4] = {v8f{}, v8f{}, v8f{}, v8f{}};

  for (int kb = 0; kb < Kpad32; kb += 32) {
    // A fragment: two aligned 16B loads (contiguous K: lane half h holds k = 16h..16h+15)
    v4u lo = {0u, 0u, 0u, 0u}, hi = {0u, 0u, 0u, 0u};
    if (coOK) {
      lo = *(const v4u*)(wrow + kb + (half << 4));
      hi = *(const v4u*)(wrow + kb + 8 + (half << 4));
    }
    struct AP { v4u a, b; } ap{lo, hi};
    const v16bf a = __builtin_bit_cast(v16bf, ap);

    // decode K -> (ci,ky,kx) once per K-step (shared by all 4 subtiles)
    int basei[16], xoff[16];
#pragma unroll
    for (int e = 0; e < 16; ++e) {
      const int k  = kb + kBhalf + e;
      const int ci = k / KS2;
      const int rr = k - ci * KS2;
      const int ky = rr / KS;
      const int kx = rr - ky * KS;
      basei[e] = (ci * KS + ky) << 6;
      xoff[e]  = (k < Ktot) ? (kx - PAD) : 1000;   // 1000 forces the bounds check to fail
    }

#pragma unroll
    for (int s = 0; s < 4; ++s) {
      const int x0 = (s << 4) + nloc;
      v16u bu;
#pragma unroll
      for (int e = 0; e < 16; ++e) {
        const int ix = x0 + xoff[e];
        bu[e] = ((unsigned)ix < 64u) ? sh[basei[e] + ix] : (unsigned short)0;
      }
      const v16bf bm = __builtin_bit_cast(v16bf, bu);
      acc[s] = __builtin_amdgcn_wmma_f32_16x16x32_bf16(
          false, a, false, bm, (short)0, acc[s], false, false);
    }
  }

  // ---- store D: element vv -> M = vv + 8*half, N = nloc ----
#pragma unroll
  for (int s = 0; s < 4; ++s) {
    const int x = (s << 4) + nloc;
#pragma unroll
    for (int vv = 0; vv < 8; ++vv) {
      const int col = mTile * 16 + vv + (half << 3);
      if (col < Cout) {
        const size_t o = (size_t)b * outBS + ((size_t)col << 12) + (y << 6) + x;
        float val = acc[s][vv] + bias[col];
        if (accumulate) val += out[o];
        out[o] = val;
      }
    }
  }
}

// ---------------- pointwise kernels ----------------
__global__ void zero_kernel(float* __restrict__ p, int n) {
  int i = blockIdx.x * 256 + threadIdx.x;
  if (i < n) p[i] = 0.0f;
}

// activations fp32 -> bf16 (compact layout)
__global__ void cvt_bf16_kernel(const float* __restrict__ in, unsigned short* __restrict__ out, int n) {
  int i = blockIdx.x * 256 + threadIdx.x;
  if (i < n) out[i] = f2bf_bits(in[i]);
}

// weights fp32 [Cout][Ktot] -> bf16 [Cout][Kpad] zero-padded
__global__ void cvt_w_kernel(const float* __restrict__ w, unsigned short* __restrict__ out,
                             int Ktot, int Kpad, int n) {
  int i = blockIdx.x * 256 + threadIdx.x;
  if (i >= n) return;
  int co = i / Kpad, k = i - co * Kpad;
  out[i] = (k < Ktot) ? f2bf_bits(w[(size_t)co * Ktot + k]) : (unsigned short)0;
}

// x[:, t] (strided) -> compact bf16 [4,5,64,64]
__global__ void cvt_x_kernel(const float* __restrict__ x, unsigned short* __restrict__ out, int t) {
  int i = blockIdx.x * 256 + threadIdx.x;
  if (i >= 4 * 5 * 4096) return;
  int b = i / (5 * 4096);
  int rem = i - b * 5 * 4096;
  out[i] = f2bf_bits(x[(size_t)b * 204800 + (size_t)t * 20480 + rem]);
}

// c_new = sigmoid(f)*c + sigmoid(i)*tanh(g)
__global__ void gates1_kernel(const float* __restrict__ xc, const float* __restrict__ hc,
                              const float* __restrict__ cc, const float* __restrict__ cold,
                              float* __restrict__ cnew) {
  int i = blockIdx.x * 256 + threadIdx.x;
  if (i >= CH_ELEMS) return;
  int pix = i & 4095, ch = (i >> 12) & 63, b = i >> 18;
  float i_t = sigmoidf_(at_blk(xc,448,0,b,ch,pix) + at_blk(hc,256,0,b,ch,pix) + at_blk(cc,192,0,b,ch,pix));
  float f_t = sigmoidf_(at_blk(xc,448,2,b,ch,pix) + at_blk(hc,256,2,b,ch,pix) + at_blk(cc,192,2,b,ch,pix) + 1.0f);
  float g_t = tanhf(at_blk(xc,448,1,b,ch,pix) + at_blk(hc,256,1,b,ch,pix) + at_blk(cc,192,1,b,ch,pix));
  cnew[i] = f_t * cold[i] + i_t * g_t;
}

// m_new = sigmoid(f')*tanh(m_m) + sigmoid(i')*tanh(g')
__global__ void gates2_kernel(const float* __restrict__ c2m, const float* __restrict__ xc,
                              const float* __restrict__ mc, float* __restrict__ mnew) {
  int i = blockIdx.x * 256 + threadIdx.x;
  if (i >= CH_ELEMS) return;
  int pix = i & 4095, ch = (i >> 12) & 63, b = i >> 18;
  float i_tp = sigmoidf_(at_blk(c2m,256,0,b,ch,pix) + at_blk(xc,448,4,b,ch,pix) + at_blk(mc,192,0,b,ch,pix));
  float f_tp = sigmoidf_(at_blk(c2m,256,2,b,ch,pix) + at_blk(xc,448,6,b,ch,pix) + at_blk(mc,192,1,b,ch,pix) + 1.0f);
  float g_tp = tanhf(at_blk(c2m,256,1,b,ch,pix) + at_blk(xc,448,5,b,ch,pix));
  mnew[i] = f_tp * tanhf(at_blk(mc,192,2,b,ch,pix)) + i_tp * g_tp;
}

// h_new = tanh(o_x+o_h+o_c+o_m) * tanh(cellconv)
__global__ void hgate_kernel(const float* __restrict__ xc, const float* __restrict__ hc,
                             const float* __restrict__ c2m, const float* __restrict__ m2o,
                             const float* __restrict__ cellc, float* __restrict__ hnew) {
  int i = blockIdx.x * 256 + threadIdx.x;
  if (i >= CH_ELEMS) return;
  int pix = i & 4095, ch = (i >> 12) & 63, b = i >> 18;
  float o_t = tanhf(at_blk(xc,448,3,b,ch,pix) + at_blk(hc,256,3,b,ch,pix) +
                    at_blk(c2m,256,3,b,ch,pix) + m2o[i]);
  hnew[i] = o_t * tanhf(cellc[i]);
}

// cat (bf16) = [c_new ; m_new]
__global__ void concat_bf16_kernel(const float* __restrict__ cn, const float* __restrict__ mn,
                                   unsigned short* __restrict__ cat) {
  int i = blockIdx.x * 256 + threadIdx.x;
  if (i >= 2 * CH_ELEMS) return;
  int pix = i & 4095, c = (i >> 12) & 127, b = i >> 19;
  float v = (c < 64) ? cn[(((size_t)b * 64 + c) << 12) + pix]
                     : mn[(((size_t)b * 64 + (c - 64)) << 12) + pix];
  cat[i] = f2bf_bits(v);
}

// z_new = u*tanh(p) + (1-u)*z
__global__ void ghu_kernel(const float* __restrict__ pz, const float* __restrict__ z,
                           float* __restrict__ zn) {
  int i = blockIdx.x * 256 + threadIdx.x;
  if (i >= CH_ELEMS) return;
  int pix = i & 4095, ch = (i >> 12) & 63, b = i >> 18;
  float pp = tanhf(at_blk(pz,128,0,b,ch,pix));
  float u  = sigmoidf_(at_blk(pz,128,1,b,ch,pix));
  zn[i] = u * pp + (1.0f - u) * z[i];
}

// ---------------- host orchestration ----------------
extern "C" void kernel_launch(void* const* d_in, const int* in_sizes, int n_in,
                              void* d_out, int out_size, void* d_ws, size_t ws_size,
                              hipStream_t stream) {
  (void)in_sizes; (void)n_in; (void)out_size; (void)ws_size;
  const float* x = (const float*)d_in[0];          // [4,10,5,64,64]
  int p = 2;                                       // d_in[1] = ls (unused)
  auto nxt = [&]() { return (const float*)d_in[p++]; };
  struct CellW { const float *wx,*bx,*wh,*bh,*wc,*bc,*wm,*bm,*wc2m,*bc2m,*wm2o,*bm2o,*wmem,*bmem; };
  CellW cw[2];
  for (int c = 0; c < 2; ++c) {
    cw[c].wx = nxt(); cw[c].bx = nxt(); cw[c].wh = nxt(); cw[c].bh = nxt();
    cw[c].wc = nxt(); cw[c].bc = nxt(); cw[c].wm = nxt(); cw[c].bm = nxt();
    cw[c].wc2m = nxt(); cw[c].bc2m = nxt(); cw[c].wm2o = nxt(); cw[c].bm2o = nxt();
    cw[c].wmem = nxt(); cw[c].bmem = nxt();
  }
  const float* g_wx = nxt(); const float* g_bx = nxt();
  const float* g_wz = nxt(); const float* g_bz = nxt();
  const float* h_w  = nxt(); const float* h_b  = nxt();

  // ---- workspace carve-up ----
  char* wsp = (char*)d_ws;
  auto allocF = [&](size_t n) { float* r = (float*)wsp; wsp += n * sizeof(float); return r; };
  auto allocW = [&](size_t n) {
    wsp = (char*)(((size_t)wsp + 15) & ~(size_t)15);
    unsigned short* r = (unsigned short*)wsp; wsp += n * 2; return r;
  };
  const size_t CH = (size_t)CH_ELEMS;
  float* xc    = allocF(7 * CH);
  float* hc    = allocF(4 * CH);
  float* cc    = allocF(3 * CH);
  float* mc    = allocF(3 * CH);
  float* c2m   = allocF(4 * CH);
  float* m2o   = allocF(CH);
  float* cellc = allocF(CH);
  float* pz    = allocF(2 * CH);
  float* st[12];
  for (int i = 0; i < 12; ++i) st[i] = allocF(CH);
  unsigned short* abf  = allocW(2 * CH);   // bf16 activation scratch (<=128 ch)
  unsigned short* catb = allocW(2 * CH);   // bf16 concat [c_new ; m_new]

  auto kp32 = [](int kt) { return (kt + 31) & ~31; };
  struct CellQ { unsigned short *wx,*wh,*wc,*wm,*wc2m,*wm2o,*wmem; };
  CellQ cq[2];
  const int cinX[2] = {5, 64};
  for (int c = 0; c < 2; ++c) {
    cq[c].wx   = allocW((size_t)448 * kp32(cinX[c] * 9));
    cq[c].wh   = allocW((size_t)256 * kp32(576));
    cq[c].wc   = allocW((size_t)192 * kp32(576));
    cq[c].wm   = allocW((size_t)192 * kp32(576));
    cq[c].wc2m = allocW((size_t)256 * kp32(576));
    cq[c].wm2o = allocW((size_t)64 * kp32(576));
    cq[c].wmem = allocW((size_t)64 * kp32(128));
  }
  unsigned short* g_wxq = allocW((size_t)128 * kp32(576));
  unsigned short* g_wzq = allocW((size_t)128 * kp32(576));
  unsigned short* h_wq  = allocW((size_t)12 * kp32(64));

  // ---- convert all weights fp32 -> padded bf16 ----
  struct Cvt { const float* s; unsigned short* d; int Cout, Ktot; };
  Cvt cv[17] = {
    {cw[0].wx, cq[0].wx, 448, 45},     {cw[0].wh, cq[0].wh, 256, 576},
    {cw[0].wc, cq[0].wc, 192, 576},    {cw[0].wm, cq[0].wm, 192, 576},
    {cw[0].wc2m, cq[0].wc2m, 256, 576},{cw[0].wm2o, cq[0].wm2o, 64, 576},
    {cw[0].wmem, cq[0].wmem, 64, 128},
    {cw[1].wx, cq[1].wx, 448, 576},    {cw[1].wh, cq[1].wh, 256, 576},
    {cw[1].wc, cq[1].wc, 192, 576},    {cw[1].wm, cq[1].wm, 192, 576},
    {cw[1].wc2m, cq[1].wc2m, 256, 576},{cw[1].wm2o, cq[1].wm2o, 64, 576},
    {cw[1].wmem, cq[1].wmem, 64, 128},
    {g_wx, g_wxq, 128, 576}, {g_wz, g_wzq, 128, 576},
    {h_w, h_wq, 12, 64},
  };
  for (int i = 0; i < 17; ++i) {
    int Kp = kp32(cv[i].Ktot), n = cv[i].Cout * Kp;
    cvt_w_kernel<<<(n + 255) / 256, 256, 0, stream>>>(cv[i].s, cv[i].d, cv[i].Ktot, Kp, n);
  }

  // ---- launchers ----
  auto conv3 = [&](const unsigned short* w, const float* bb, const unsigned short* inb,
                   int Cin, int Cout, float* out, int acc) {
    int Ktot = Cin * 9;
    dim3 g((unsigned)((Cout + 63) / 64), 256);
    conv_wmma_kernel<3><<<g, 128, 0, stream>>>(w, bb, inb, out, Cout * 4096,
                                               Cin, Cout, Ktot, kp32(Ktot), acc);
  };
  auto conv1 = [&](const unsigned short* w, const float* bb, const unsigned short* inb,
                   int Cin, int Cout, float* out, int acc) {
    dim3 g((unsigned)((Cout + 63) / 64), 256);
    conv_wmma_kernel<1><<<g, 128, 0, stream>>>(w, bb, inb, out, Cout * 4096,
                                               Cin, Cout, Cin, kp32(Cin), acc);
  };
  const int GP = (int)(CH / 256);
  auto cvtAct = [&](const float* src, int C) {
    int n = 4 * C * 4096;
    cvt_bf16_kernel<<<(n + 255) / 256, 256, 0, stream>>>(src, abf, n);
  };

  // ---- state pointers (ping-pong) ----
  float *h0 = st[0], *h0n = st[1], *c0s = st[2], *c0n = st[3];
  float *h1 = st[4], *h1n = st[5], *c1s = st[6], *c1n = st[7];
  float *m  = st[8], *mn  = st[9], *z   = st[10], *zn  = st[11];
  float* zi[6] = {h0, c0s, h1, c1s, m, z};
  for (int i = 0; i < 6; ++i) zero_kernel<<<GP, 256, 0, stream>>>(zi[i], (int)CH);

  auto runCell = [&](const CellQ& Q, const CellW& W, int CinX,
                     float* h, float* c, float* mcur, float* hn, float* cn, float* mnew) {
    // abf already holds this cell's x-input (CinX channels, bf16)
    conv3(Q.wx, W.bx, abf, CinX, 448, xc, 0);
    cvtAct(h, 64);    conv3(Q.wh, W.bh, abf, 64, 256, hc, 0);
    cvtAct(c, 64);    conv3(Q.wc, W.bc, abf, 64, 192, cc, 0);
    cvtAct(mcur, 64); conv3(Q.wm, W.bm, abf, 64, 192, mc, 0);
    gates1_kernel<<<GP, 256, 0, stream>>>(xc, hc, cc, c, cn);
    cvtAct(cn, 64);   conv3(Q.wc2m, W.bc2m, abf, 64, 256, c2m, 0);
    gates2_kernel<<<GP, 256, 0, stream>>>(c2m, xc, mc, mnew);
    cvtAct(mnew, 64); conv3(Q.wm2o, W.bm2o, abf, 64, 64, m2o, 0);
    concat_bf16_kernel<<<2 * GP, 256, 0, stream>>>(cn, mnew, catb);
    conv1(Q.wmem, W.bmem, catb, 128, 64, cellc, 0);
    hgate_kernel<<<GP, 256, 0, stream>>>(xc, hc, c2m, m2o, cellc, hn);
  };

  for (int t = 0; t < 10; ++t) {
    cvt_x_kernel<<<(4 * 5 * 4096 + 255) / 256, 256, 0, stream>>>(x, abf, t);
    runCell(cq[0], cw[0], 5, h0, c0s, m, h0n, c0n, mn);
    std::swap(h0, h0n); std::swap(c0s, c0n); std::swap(m, mn);

    cvtAct(h0, 64); conv3(g_wxq, g_bx, abf, 64, 128, pz, 0);
    cvtAct(z, 64);  conv3(g_wzq, g_bz, abf, 64, 128, pz, 1);   // accumulate
    ghu_kernel<<<GP, 256, 0, stream>>>(pz, z, zn);
    std::swap(z, zn);

    cvtAct(z, 64);
    runCell(cq[1], cw[1], 64, h1, c1s, m, h1n, c1n, mn);
    std::swap(h1, h1n); std::swap(c1s, c1n); std::swap(m, mn);
  }

  // head: 1x1 conv 64 -> 12, writes d_out [4,12,64,64]
  cvtAct(h1, 64);
  conv1(h_wq, h_b, abf, 64, 12, (float*)d_out, 0);
}
